// ScoreNet_18829136625753
// MI455X (gfx1250) — matrix operand order repaired and checked by reading, not compiled, not run on hardware
//
#include <hip/hip_runtime.h>
#include <hip/hip_bf16.h>

// ---------------------------------------------------------------------------
// DGCNN ScoreNet on MI455X (gfx1250), wave32 + v_wmma_f32_16x16x32_bf16.
// EdgeConv factorized:  y[p,k,:] = A[idx[p,k],:] + Bm[p,:]   where
//   A  = h @ Wn^T,  Bm = h @ (Wc-Wn)^T   -> one stacked GEMM h @ [Wn;Wc-Wn]^T
// max over k commutes with BN(s>0)+LeakyReLU (monotone), applied post-max.
// KNN fused: 16x1024 strip of shifted pd computed by WMMA into LDS, top-20
// selected in-kernel via wave32 shuffle argmax.
// WMMA loops are fully unrolled + software-pipelined (next-tile B fragments
// prefetched before the current tile's WMMAs) so loads issued one tile ahead
// cover the s_wait_loadcnt before each v_wmma.
// ---------------------------------------------------------------------------

typedef __attribute__((ext_vector_type(16))) __bf16 v16bf;
typedef __attribute__((ext_vector_type(8)))  float  v8f;

#define BPts   16384      // B*P = 16*1024
#define PPTS   1024
#define NCLU   16

__device__ __forceinline__ unsigned short f2bf(float f) {
  union { float f; unsigned u; } x; x.f = f;
  unsigned u = x.u;
  u += 0x7FFFu + ((u >> 16) & 1u);          // round-to-nearest-even
  return (unsigned short)(u >> 16);
}

union Frag { v16bf v; uint4 q[2]; };

// A fragment 16x32 (MxK): lanes 0-15 hold K {0..7,16..23}, lanes 16-31 hold
// K {8..15,24..31} of row M = mBase + (lane&15).      (ISA 7.12.2)
__device__ __forceinline__ v16bf load_frag_a(const unsigned short* A, int lda,
                                             int mBase, int kBase) {
  int lane = threadIdx.x & 31;
  int row  = mBase + (lane & 15);
  int k0   = kBase + ((lane >> 4) << 3);
  Frag f;
  f.q[0] = *(const uint4*)(A + (size_t)row * lda + k0);
  f.q[1] = *(const uint4*)(A + (size_t)row * lda + k0 + 16);
  return f.v;
}

// B fragment 32x16 (KxN), B stored row-major [N, K] (K contiguous):
// lanes 0-15 hold K {0..7,16..23}, lanes 16-31 hold K {8..15,24..31}
// of col N = nBase + (lane&15) (same K order as A).
__device__ __forceinline__ v16bf load_frag_b(const unsigned short* B, int ldb,
                                             int nBase, int kBase) {
  int lane = threadIdx.x & 31;
  int col  = nBase + (lane & 15);
  int k0   = kBase + ((lane >> 4) << 3);
  const uint4* p = (const uint4*)(B + (size_t)col * ldb + k0);
  Frag f; f.q[0] = p[0]; f.q[1] = p[2];
  return f.v;
}

__device__ __forceinline__ v8f wmma_bf16(v16bf a, v16bf b, v8f c) {
  return __builtin_amdgcn_wmma_f32_16x16x32_bf16(
      /*neg_a=*/false, a, /*neg_b=*/false, b,
      /*c_mod=*/(short)0, c, /*reuse_a=*/false, /*reuse_b=*/false);
}

// ---------------------------------------------------------------------------
// Weight prep: out[r,c] = r<O ? W[r, c] : W[r-O, C+c] - W[r-O, c]   (bf16)
__global__ void stack_weights(const float* __restrict__ W, int O, int C,
                              unsigned short* __restrict__ out) {
  int r = blockIdx.x;           // 0 .. 2O-1
  int c = threadIdx.x;          // 0 .. C-1
  float v;
  if (r < O) v = W[(size_t)r * (2 * C) + c];
  else       v = W[(size_t)(r - O) * (2 * C) + C + c]
               - W[(size_t)(r - O) * (2 * C) + c];
  out[(size_t)r * C + c] = f2bf(v);
}

__global__ void cast_bf(const float* __restrict__ in,
                        unsigned short* __restrict__ out, int n) {
  int i = blockIdx.x * blockDim.x + threadIdx.x;
  if (i < n) out[i] = f2bf(in[i]);
}

// Gather cluster points -> bf16 features + squared norms (one wave / point).
__global__ void gather_feats(const float* __restrict__ feats,
                             const int* __restrict__ clusters,
                             unsigned short* __restrict__ h0b,
                             float* __restrict__ sq) {
  int p    = blockIdx.x;        // 0..16383
  int lane = threadIdx.x;       // 32
  int src  = clusters[p];
  const float* f = feats + (size_t)src * 128;
  float a = 0.f;
  #pragma unroll
  for (int c = lane; c < 128; c += 32) {
    float v = f[c];
    h0b[(size_t)p * 128 + c] = f2bf(v);
    a += v * v;
  }
  #pragma unroll
  for (int off = 16; off > 0; off >>= 1) a += __shfl_xor(a, off, 32);
  if (lane == 0) sq[p] = a;
}

// Squared norms of a column slice of the f32 concat buffer.
__global__ void sqcols(const float* __restrict__ xc, int ld, int colOff, int C,
                       float* __restrict__ sq) {
  int p = blockIdx.x * blockDim.x + threadIdx.x;
  if (p >= BPts) return;
  const float* row = xc + (size_t)p * ld + colOff;
  float a = 0.f;
  for (int c = 0; c < C; c++) { float v = row[c]; a += v * v; }
  sq[p] = a;
}

// ---------------------------------------------------------------------------
// Fused KNN: per block (grid: 64 x 1 x 16) compute a 16x1024 strip of
// pd[i,j] = 2*(h_i . h_j) - |h_j|^2  (row-constant -|h_i|^2 dropped: ordering
// invariant) via WMMA into LDS, then top-20 per row (self included, like ref).
template <int C>
__global__ void knn_topk(const unsigned short* __restrict__ Hb, int lda,
                         const float* __restrict__ sq,
                         int* __restrict__ knn_idx) {
  extern __shared__ float pd[];                 // [16][1024] = 64 KB
  constexpr int NKT = C / 32;                   // K-steps (2 or 4)
  int z     = blockIdx.z;
  int mBase = blockIdx.x * 16;
  int wave  = threadIdx.x >> 5;
  int lane  = threadIdx.x & 31;
  const unsigned short* H = Hb + (size_t)z * PPTS * lda;
  const float* sqz = sq + z * PPTS;

  v8f zero = {0.f,0.f,0.f,0.f,0.f,0.f,0.f,0.f};
  v16bf afr[NKT];                               // statically indexed
  #pragma unroll
  for (int kt = 0; kt < NKT; kt++) afr[kt] = load_frag_a(H, lda, mBase, kt * 32);

  // 4 waves, each covers 16 n-tiles (256 columns); B frags double-buffered.
  int nBase0 = wave * 256;
  v16bf bcur[NKT], bnxt[NKT];
  #pragma unroll
  for (int kt = 0; kt < NKT; kt++) bcur[kt] = load_frag_b(H, lda, nBase0, kt * 32);
  #pragma unroll
  for (int t = 0; t < 16; t++) {
    int nBase = nBase0 + t * 16;
    if (t < 15) {                               // prefetch next tile
      #pragma unroll
      for (int kt = 0; kt < NKT; kt++)
        bnxt[kt] = load_frag_b(H, lda, nBase + 16, kt * 32);
    }
    v8f acc = zero;
    #pragma unroll
    for (int kt = 0; kt < NKT; kt++) acc = wmma_bf16(afr[kt], bcur[kt], acc);
    int mrow = (lane >> 4) << 3;                // D layout: M = v + 8*(lane>=16)
    int col  = nBase + (lane & 15);
    float sj = sqz[col];
    #pragma unroll
    for (int v = 0; v < 8; v++)
      pd[(mrow + v) * PPTS + col] = 2.f * acc[v] - sj;
    #pragma unroll
    for (int kt = 0; kt < NKT; kt++) bcur[kt] = bnxt[kt];  // renamed away
  }
  __syncthreads();

  // top-20 per row; wave handles 4 rows; shuffle argmax, ties -> lower index
  for (int r = 0; r < 4; r++) {
    int row = wave * 4 + r;
    float* prow = pd + row * PPTS;
    int* outp = knn_idx + ((size_t)(z * PPTS + mBase + row)) * 20;
    for (int kk = 0; kk < 20; kk++) {
      float best = -3.0e38f; int bidx = 0;
      #pragma unroll 8
      for (int j = lane; j < PPTS; j += 32) {
        float vv = prow[j];
        if (vv > best) { best = vv; bidx = j; }
      }
      #pragma unroll
      for (int off = 16; off > 0; off >>= 1) {
        float ov = __shfl_xor(best, off, 32);
        int   oi = __shfl_xor(bidx, off, 32);
        if (ov > best || (ov == best && oi < bidx)) { best = ov; bidx = oi; }
      }
      if (lane == 0) outp[kk] = bidx;
      if ((bidx & 31) == lane) prow[bidx] = -3.0e38f;  // knock out winner
    }
  }
}

// ---------------------------------------------------------------------------
// C[M,N] = A[M,KD](bf16) @ Bw[N,KD]^T(bf16), f32 accum. Block = 4 waves, each
// wave owns a 16x64 strip (A fragment reused across 4 WMMAs per K-step).
// K-loop fully unrolled with one-step-ahead prefetch of A and B fragments.
template <int KD>
__global__ void gemm_bt(const unsigned short* __restrict__ A, int lda,
                        const unsigned short* __restrict__ Bw, int ldb,
                        float* __restrict__ C, int ldc) {
  int wave  = threadIdx.x >> 5;
  int lane  = threadIdx.x & 31;
  int mBase = (blockIdx.x * 4 + wave) * 16;
  int nBase = blockIdx.y * 64;
  v8f zero = {0.f,0.f,0.f,0.f,0.f,0.f,0.f,0.f};
  v8f acc[4] = {zero, zero, zero, zero};

  v16bf a0 = load_frag_a(A, lda, mBase, 0);
  v16bf b0[4];
  #pragma unroll
  for (int s = 0; s < 4; s++) b0[s] = load_frag_b(Bw, ldb, nBase + s * 16, 0);

  #pragma unroll
  for (int kt = 0; kt < KD; kt += 32) {
    v16bf a1 = a0;
    v16bf b1[4] = {b0[0], b0[1], b0[2], b0[3]};
    if (kt + 32 < KD) {                         // prefetch next K-step
      a1 = load_frag_a(A, lda, mBase, kt + 32);
      #pragma unroll
      for (int s = 0; s < 4; s++)
        b1[s] = load_frag_b(Bw, ldb, nBase + s * 16, kt + 32);
    }
    #pragma unroll
    for (int s = 0; s < 4; s++) acc[s] = wmma_bf16(a0, b0[s], acc[s]);
    a0 = a1;
    #pragma unroll
    for (int s = 0; s < 4; s++) b0[s] = b1[s];  // renamed away (full unroll)
  }

  int mrow = (lane >> 4) << 3;
  int ncol = lane & 15;
  #pragma unroll
  for (int s = 0; s < 4; s++)
    #pragma unroll
    for (int v = 0; v < 8; v++)
      C[(size_t)(mBase + mrow + v) * ldc + nBase + s * 16 + ncol] = acc[s][v];
}

// out[p,o] = lrelu(s*(max_k A[idx[p,k],o] + Bm[p,o]) + b); writes f32 slice of
// the concat buffer and its bf16 mirror. AB = [A | Bm], ldc = 2*O.
__global__ void maxgather(const float* __restrict__ AB, int ldc, int O,
                          const int* __restrict__ knn_idx,
                          const float* __restrict__ s, const float* __restrict__ b,
                          float* __restrict__ xc, unsigned short* __restrict__ xcb,
                          int colOff) {
  int p = blockIdx.x;                 // 0..16383
  int o = threadIdx.x;                // 0..O-1
  int z = p >> 10;
  const int* nb = knn_idx + (size_t)p * 20;
  float m = -3.0e38f;
  #pragma unroll 4
  for (int k = 0; k < 20; k++) {
    int j = z * PPTS + nb[k];
    m = fmaxf(m, AB[(size_t)j * ldc + o]);
  }
  float val = m + AB[(size_t)p * ldc + O + o];
  val = val * s[o] + b[o];
  val = (val > 0.f) ? val : 0.2f * val;
  size_t outi = (size_t)p * 512 + colOff + o;
  xc[outi]  = val;
  xcb[outi] = f2bf(val);
}

// conv5 BN+lrelu fused with max+avg pooling over the cluster.
__global__ void pool_conv5(const float* __restrict__ Y,    // [16384,512]
                           const float* __restrict__ s5, const float* __restrict__ b5,
                           float* __restrict__ feat) {     // [16,1024]
  int z = blockIdx.x;
  int o = threadIdx.x;                // 512
  float sc = s5[o], sh = b5[o];
  float mx = -3.0e38f, sm = 0.f;
  const float* base = Y + (size_t)z * PPTS * 512 + o;
  #pragma unroll 4
  for (int p = 0; p < PPTS; p++) {
    float v = base[(size_t)p * 512] * sc + sh;
    v = (v > 0.f) ? v : 0.2f * v;
    mx = fmaxf(mx, v); sm += v;
  }
  feat[z * 1024 + o]       = mx;
  feat[z * 1024 + 512 + o] = sm * (1.f / 1024.f);
}

// Tiny MLP head (16 rows total — scalar f32 is the right tool here).
__global__ void head(const float* __restrict__ feat,
                     const float* __restrict__ L1, const float* __restrict__ s6,
                     const float* __restrict__ b6,
                     const float* __restrict__ L2, const float* __restrict__ bl2,
                     const float* __restrict__ s7, const float* __restrict__ b7,
                     const float* __restrict__ L3, const float* __restrict__ bl3,
                     float* __restrict__ out) {
  __shared__ float z1[512];
  __shared__ float z2[256];
  int z = blockIdx.x, t = threadIdx.x;     // 512 threads
  const float* f = feat + z * 1024;
  float a = 0.f;
  const float* w = L1 + (size_t)t * 1024;
  #pragma unroll 4
  for (int k = 0; k < 1024; k++) a += f[k] * w[k];
  a = a * s6[t] + b6[t];
  z1[t] = (a > 0.f) ? a : 0.2f * a;
  __syncthreads();
  if (t < 256) {
    float c = 0.f;
    const float* w2 = L2 + (size_t)t * 512;
    #pragma unroll 4
    for (int k = 0; k < 512; k++) c += z1[k] * w2[k];
    c = (c + bl2[t]) * s7[t] + b7[t];
    z2[t] = (c > 0.f) ? c : 0.2f * c;
  }
  __syncthreads();
  if (t == 0) {
    float c = 0.f;
    for (int k = 0; k < 256; k++) c += z2[k] * L3[k];
    c += bl3[0];
    out[z] = 1.f / (1.f + __expf(-c));
  }
}

// ---------------------------------------------------------------------------
extern "C" void kernel_launch(void* const* d_in, const int* in_sizes, int n_in,
                              void* d_out, int out_size, void* d_ws, size_t ws_size,
                              hipStream_t stream) {
  const float* feats    = (const float*)d_in[0];
  const int*   clusters = (const int*)  d_in[1];
  const float* W1 = (const float*)d_in[2];  const float* s1 = (const float*)d_in[3];  const float* b1 = (const float*)d_in[4];
  const float* W2 = (const float*)d_in[5];  const float* s2 = (const float*)d_in[6];  const float* b2 = (const float*)d_in[7];
  const float* W3 = (const float*)d_in[8];  const float* s3 = (const float*)d_in[9];  const float* b3 = (const float*)d_in[10];
  const float* W4 = (const float*)d_in[11]; const float* s4 = (const float*)d_in[12]; const float* b4 = (const float*)d_in[13];
  const float* W5 = (const float*)d_in[14]; const float* s5 = (const float*)d_in[15]; const float* b5 = (const float*)d_in[16];
  const float* L1 = (const float*)d_in[17]; const float* s6 = (const float*)d_in[18]; const float* b6 = (const float*)d_in[19];
  const float* L2 = (const float*)d_in[20]; const float* bl2= (const float*)d_in[21];
  const float* s7 = (const float*)d_in[22]; const float* b7 = (const float*)d_in[23];
  const float* L3 = (const float*)d_in[24]; const float* bl3= (const float*)d_in[25];
  float* out = (float*)d_out;

  char* wsp = (char*)d_ws;
  auto alloc = [&](size_t bytes) -> void* {
    void* p = wsp; wsp += (bytes + 255) & ~(size_t)255; return p;
  };
  unsigned short* Wstk1 = (unsigned short*)alloc(128 * 128 * 2);
  unsigned short* Wstk2 = (unsigned short*)alloc(128 * 64  * 2);
  unsigned short* Wstk3 = (unsigned short*)alloc(256 * 64  * 2);
  unsigned short* Wstk4 = (unsigned short*)alloc(512 * 128 * 2);
  unsigned short* W5b   = (unsigned short*)alloc(512 * 512 * 2);
  unsigned short* h0b   = (unsigned short*)alloc((size_t)BPts * 128 * 2);
  unsigned short* xcb   = (unsigned short*)alloc((size_t)BPts * 512 * 2);
  float*          xc    = (float*)alloc((size_t)BPts * 512 * 4);
  float*          AB    = (float*)alloc((size_t)BPts * 512 * 4);
  float*          sq    = (float*)alloc((size_t)BPts * 4);
  int*            kidx  = (int*)  alloc((size_t)BPts * 20 * 4);
  float*          feat  = (float*)alloc((size_t)NCLU * 1024 * 4);

  // ---- weight prep (bf16 stacked [Wn ; Wc-Wn]) ----
  stack_weights<<<dim3(128), dim3(128), 0, stream>>>(W1,  64, 128, Wstk1);
  stack_weights<<<dim3(128), dim3( 64), 0, stream>>>(W2,  64,  64, Wstk2);
  stack_weights<<<dim3(256), dim3( 64), 0, stream>>>(W3, 128,  64, Wstk3);
  stack_weights<<<dim3(512), dim3(128), 0, stream>>>(W4, 256, 128, Wstk4);
  cast_bf<<<dim3((512 * 512 + 255) / 256), dim3(256), 0, stream>>>(W5, W5b, 512 * 512);

  // ---- gather cluster features ----
  gather_feats<<<dim3(BPts), dim3(32), 0, stream>>>(feats, clusters, h0b, sq);

  const size_t pdBytes = 16 * PPTS * sizeof(float);   // 64 KB LDS strip
  dim3 knnGrid(PPTS / 16, 1, NCLU);
  dim3 gemmBlk(128);

  // ---- EdgeConv 1: C=128 -> O=64, writes xc[:,0:64] ----
  knn_topk<128><<<knnGrid, dim3(128), pdBytes, stream>>>(h0b, 128, sq, kidx);
  gemm_bt<128><<<dim3(BPts / 64, 2), gemmBlk, 0, stream>>>(h0b, 128, Wstk1, 128, AB, 128);
  maxgather<<<dim3(BPts), dim3(64), 0, stream>>>(AB, 128, 64, kidx, s1, b1, xc, xcb, 0);
  sqcols<<<dim3(BPts / 256), dim3(256), 0, stream>>>(xc, 512, 0, 64, sq);

  // ---- EdgeConv 2: C=64 -> O=64, writes xc[:,64:128] ----
  knn_topk<64><<<knnGrid, dim3(128), pdBytes, stream>>>(xcb, 512, sq, kidx);
  gemm_bt<64><<<dim3(BPts / 64, 2), gemmBlk, 0, stream>>>(xcb, 512, Wstk2, 64, AB, 128);
  maxgather<<<dim3(BPts), dim3(64), 0, stream>>>(AB, 128, 64, kidx, s2, b2, xc, xcb, 64);
  sqcols<<<dim3(BPts / 256), dim3(256), 0, stream>>>(xc, 512, 64, 64, sq);

  // ---- EdgeConv 3: C=64 -> O=128, writes xc[:,128:256] ----
  knn_topk<64><<<knnGrid, dim3(128), pdBytes, stream>>>(xcb + 64, 512, sq, kidx);
  gemm_bt<64><<<dim3(BPts / 64, 4), gemmBlk, 0, stream>>>(xcb + 64, 512, Wstk3, 64, AB, 256);
  maxgather<<<dim3(BPts), dim3(128), 0, stream>>>(AB, 256, 128, kidx, s3, b3, xc, xcb, 128);
  sqcols<<<dim3(BPts / 256), dim3(256), 0, stream>>>(xc, 512, 128, 128, sq);

  // ---- EdgeConv 4: C=128 -> O=256, writes xc[:,256:512] ----
  knn_topk<128><<<knnGrid, dim3(128), pdBytes, stream>>>(xcb + 128, 512, sq, kidx);
  gemm_bt<128><<<dim3(BPts / 64, 8), gemmBlk, 0, stream>>>(xcb + 128, 512, Wstk4, 128, AB, 512);
  maxgather<<<dim3(BPts), dim3(256), 0, stream>>>(AB, 512, 256, kidx, s4, b4, xc, xcb, 256);

  // ---- conv5: [16384,512] @ W5^T -> AB, then fused BN+lrelu+pool ----
  gemm_bt<512><<<dim3(BPts / 64, 8), gemmBlk, 0, stream>>>(xcb, 512, W5b, 512, AB, 512);
  pool_conv5<<<dim3(NCLU), dim3(512), 0, stream>>>(AB, s5, b5, feat);

  // ---- MLP head + sigmoid ----
  head<<<dim3(NCLU), dim3(512), 0, stream>>>(feat, L1, s6, b6, L2, bl2, s7, b7, L3, bl3, out);
  (void)in_sizes; (void)n_in; (void)out_size; (void)ws_size;
}
